// Attention_8280696946966
// MI455X (gfx1250) — matrix operand order repaired and checked by reading, not compiled, not run on hardware
//
#include <hip/hip_runtime.h>
#include <hip/hip_bf16.h>

// ---------------------------------------------------------------------------
// Causal multi-head attention block for MI455X (gfx1250, wave32, WMMA + TDM).
//   DIM=2048, H=16, HD=128, B=2, S=2048.
// Compute-bound (~206 GFLOP vs ~100MB unique data @ 23.3 TB/s) ->
//   f16 WMMA w/ f32 accumulate; GEMMs fed by Tensor-Data-Mover async loads
//   into LDS (double-buffered, TENSORcnt-tracked).
// ---------------------------------------------------------------------------

typedef _Float16 half_t;
typedef __attribute__((ext_vector_type(16))) _Float16     v16h;
typedef __attribute__((ext_vector_type(8)))  float        v8f;
typedef __attribute__((ext_vector_type(4)))  unsigned int v4u;
typedef __attribute__((ext_vector_type(8)))  unsigned int v8u;

#define DIMX   2048
#define NHEADS 16
#define HD     128
#define BSZ    2
#define SEQ    2048
#define NTOK   (BSZ * SEQ)
#define RHALF  (HD / 2)

#define NEGBIG  -3.0e38f
#define MASKNEG -1.0e9f

// GEMM blocking
#define MBLK 128
#define NBLK 64
#define KBLK 64

union F16x16 { v16h v; uint4 u[2]; };
union F32x8  { v8f  v; float f[8]; };

// Generic->LDS offset: per CDNA5 ISA 10.2, LDS generic addresses hold the LDS
// byte offset in addr[31:0], so truncation yields the D#.lds_addr value.
__device__ __forceinline__ unsigned int lds_offset_of(const void* p) {
  return (unsigned int)(uintptr_t)p;
}

// ---------------------------------------------------------------------------
// Tensor Data Mover: async 2D tile load Global -> LDS (D# groups 0/1 only,
// VADDR2/3 = NULL -> 2D). Issued per-wave (EXEC ignored); completion tracked
// with TENSORcnt. Inline asm form is portable across ROCm7.2 / clang-23.
// ---------------------------------------------------------------------------
__device__ __forceinline__ void tdm_load_2d_f16(unsigned int  lds_byte_addr,
                                                const half_t* gaddr,
                                                unsigned int  tensor_w,   // elems
                                                unsigned int  tensor_h,   // rows
                                                unsigned int  row_stride, // elems
                                                unsigned int  tile_w,     // elems
                                                unsigned int  tile_h) {   // rows
  const unsigned long long ga = (unsigned long long)(uintptr_t)gaddr;
  v4u g0;
  g0.x = 0x1u;                                   // count=1, user desc, no gather
  g0.y = lds_byte_addr;                          // [63:32]  lds_addr
  g0.z = (unsigned int)ga;                       // [95:64]  global_addr lo
  g0.w = ((unsigned int)(ga >> 32) & 0x01FFFFFFu) | 0x80000000u; // type=2
  v8u g1;
  g1.s0 = (1u << 16);                            // data_size=1 (2 bytes)
  g1.s1 = (tensor_w & 0xFFFFu) << 16;            // tensor_dim0[15:0] @ [63:48]
  g1.s2 = (tensor_w >> 16) | ((tensor_h & 0xFFFFu) << 16); // dim0 hi | dim1 lo
  g1.s3 = (tensor_h >> 16) | (tile_w << 16);     // dim1 hi | tile_dim0
  g1.s4 = tile_h;                                // tile_dim1 (tile_dim2=0)
  g1.s5 = row_stride;                            // tensor_dim0_stride[31:0]
  g1.s6 = 0u;
  g1.s7 = 0u;
  asm volatile("tensor_load_to_lds %0, %1" :: "s"(g0), "s"(g1) : "memory");
}

// Row reductions confined to 16-lane halves of the wave32.
__device__ __forceinline__ float row_max16(float x) {
#pragma unroll
  for (int m = 1; m < 16; m <<= 1) x = fmaxf(x, __shfl_xor(x, m, 32));
  return x;
}
__device__ __forceinline__ float row_sum16(float x) {
#pragma unroll
  for (int m = 1; m < 16; m <<= 1) x += __shfl_xor(x, m, 32);
  return x;
}

// ---------------------------------------------------------------------------
// Elementwise f32 -> f16 conversion (grid-stride).
// ---------------------------------------------------------------------------
__global__ void cvt_f32_f16(const float* __restrict__ in, half_t* __restrict__ out, int n) {
  int i = blockIdx.x * blockDim.x + threadIdx.x;
  int stride = gridDim.x * blockDim.x;
  for (; i < n; i += stride) out[i] = (half_t)in[i];
}

// ---------------------------------------------------------------------------
// Block-tiled NT GEMM: C[M,N] = A[M,K] * B[N,K]^T, f16 in / f32 accumulate.
// 256 threads = 8 waves (4 along M x 2 along N); block tile 128x64; each wave
// owns a 32x32 register tile (2x2 WMMA accumulators, 2x fragment reuse).
// K-step 64, TDM double-buffered: wave0 issues tensor_load_to_lds for the
// next A/B tiles, waits TENSORcnt<=2 so the next buffer's DMAs overlap the
// current buffer's WMMAs. Fragments are read from LDS with ds_load_b128 in
// the ISA-mandated wave32 A/B layouts.
// ---------------------------------------------------------------------------
__global__ void __launch_bounds__(256)
gemm_tdm_nt(const half_t* __restrict__ A, const half_t* __restrict__ Bm,
            float* __restrict__ Cf, half_t* __restrict__ Ch,
            int M, int N, int K) {
  __shared__ __align__(16) half_t As[2][MBLK][KBLK];   // 32 KB
  __shared__ __align__(16) half_t Bs[2][NBLK][KBLK];   // 16 KB

  const int tid  = threadIdx.x;
  const int lane = tid & 31;
  const int ln   = lane & 15;
  const int hi   = lane >> 4;
  const int w    = tid >> 5;          // wave id 0..7
  const int wm   = (w & 3) * 32;      // wave M offset in block
  const int wn   = (w >> 2) * 32;     // wave N offset in block

  const half_t* Ablk = A  + (size_t)(blockIdx.y * MBLK) * K;
  const half_t* Bblk = Bm + (size_t)(blockIdx.x * NBLK) * K;

  const int nkb = K / KBLK;

  // Prologue: DMA the first K-slab into buffer 0 (wave 0 only; EXEC ignored
  // by TDM, branch keeps the other waves from issuing duplicates).
  if (tid < 32) {
    tdm_load_2d_f16(lds_offset_of(&As[0][0][0]), Ablk, K, M, K, KBLK, MBLK);
    tdm_load_2d_f16(lds_offset_of(&Bs[0][0][0]), Bblk, K, N, K, KBLK, NBLK);
  }

  F32x8 acc[2][2] = {};

  for (int kb = 0; kb < nkb; ++kb) {
    const int cur = kb & 1;
    const int nxt = cur ^ 1;

    __syncthreads();   // everyone done reading buffer 'nxt' from iter kb-1
    if (tid < 32) {
      if (kb + 1 < nkb) {
        const int k0 = (kb + 1) * KBLK;
        tdm_load_2d_f16(lds_offset_of(&As[nxt][0][0]), Ablk + k0, K, M, K, KBLK, MBLK);
        tdm_load_2d_f16(lds_offset_of(&Bs[nxt][0][0]), Bblk + k0, K, N, K, KBLK, NBLK);
        __builtin_amdgcn_s_wait_tensorcnt(2);  // 'cur' tiles landed; 'nxt' in flight
      } else {
        __builtin_amdgcn_s_wait_tensorcnt(0);
      }
    }
    __syncthreads();   // buffer 'cur' visible to all waves

#pragma unroll
    for (int ks = 0; ks < KBLK / 32; ++ks) {
      F16x16 af[2], bf[2];
#pragma unroll
      for (int mi = 0; mi < 2; ++mi) {   // A frag: lane = M row, VGPRs hold K
        const half_t* ar = &As[cur][wm + mi * 16 + ln][ks * 32];
        af[mi].u[0] = *(const uint4*)(ar + hi * 8);
        af[mi].u[1] = *(const uint4*)(ar + 16 + hi * 8);
      }
#pragma unroll
      for (int ni = 0; ni < 2; ++ni) {   // B frag: lane = N col, VGPRs hold K
        const half_t* br = &Bs[cur][wn + ni * 16 + ln][ks * 32];
        bf[ni].u[0] = *(const uint4*)(br + hi * 8);
        bf[ni].u[1] = *(const uint4*)(br + 16 + hi * 8);
      }
#pragma unroll
      for (int mi = 0; mi < 2; ++mi)
#pragma unroll
        for (int ni = 0; ni < 2; ++ni)
          acc[mi][ni].v = __builtin_amdgcn_wmma_f32_16x16x32_f16(
              false, af[mi].v, false, bf[ni].v, (short)0, acc[mi][ni].v,
              false, false);
    }
  }

  // Epilogue: C layout row = v + 8*hi, col = lane&15.
  const int gm0 = blockIdx.y * MBLK + wm;
  const int gn0 = blockIdx.x * NBLK + wn;
#pragma unroll
  for (int mi = 0; mi < 2; ++mi)
#pragma unroll
    for (int ni = 0; ni < 2; ++ni) {
      const int cn = gn0 + ni * 16 + ln;
#pragma unroll
      for (int v = 0; v < 8; ++v) {
        const int cm = gm0 + mi * 16 + v + 8 * hi;
        if (Ch) Ch[(size_t)cm * N + cn] = (half_t)acc[mi][ni].f[v];
        else    Cf[(size_t)cm * N + cn] = acc[mi][ni].f[v];
      }
    }
}

// ---------------------------------------------------------------------------
// RoPE + repack (NTOK,DIM) -> (B,H,S,HD), f16.
// ---------------------------------------------------------------------------
__global__ void rope_pack(const half_t* __restrict__ T,
                          const float* __restrict__ fcos,
                          const float* __restrict__ fsin,
                          half_t* __restrict__ Out) {
  const int total = NTOK * DIMX / 2;
  int gid = blockIdx.x * blockDim.x + threadIdx.x;
  if (gid >= total) return;
  const int i = gid % RHALF;
  const int h = (gid / RHALF) % NHEADS;
  const int t = gid / (RHALF * NHEADS);
  const int b = t / SEQ, s = t % SEQ;

  const float a  = (float)T[(size_t)t * DIMX + h * HD + 2 * i];
  const float bb = (float)T[(size_t)t * DIMX + h * HD + 2 * i + 1];
  const float c  = fcos[s * RHALF + i];
  const float sn = fsin[s * RHALF + i];

  const size_t o = (((size_t)b * NHEADS + h) * SEQ + s) * HD + 2 * i;
  Out[o]     = (half_t)(a * c - bb * sn);
  Out[o + 1] = (half_t)(a * sn + bb * c);
}

// ---------------------------------------------------------------------------
// V repack (NTOK,DIM) -> (B,H,HD,S)  (transposed so PV B-fragments are
// contiguous along the key dimension).
// ---------------------------------------------------------------------------
__global__ void v_pack(const half_t* __restrict__ T, half_t* __restrict__ Vt) {
  const int total = NTOK * DIMX;
  int gid = blockIdx.x * blockDim.x + threadIdx.x;
  if (gid >= total) return;
  const int d = gid % HD;
  const int h = (gid / HD) % NHEADS;
  const int t = gid / (HD * NHEADS);
  const int b = t / SEQ, s = t % SEQ;
  Vt[(((size_t)b * NHEADS + h) * HD + d) * SEQ + s] =
      T[(size_t)t * DIMX + h * HD + d];
}

// ---------------------------------------------------------------------------
// Flash-style causal attention. One wave32 per (b, h, 16-query tile).
// Key loop steps 32 keys: 8 WMMAs for scores (2 key tiles x 4 d-chunks),
// online softmax in f32, P transposed C-layout -> A-layout through LDS,
// then 8 WMMAs for P*V (8 d-tiles of 16). Epilogue scales by 1/l.
// ---------------------------------------------------------------------------
__global__ void __launch_bounds__(32)
attn_fwd(const half_t* __restrict__ Qh, const half_t* __restrict__ Kh,
         const half_t* __restrict__ Vt, half_t* __restrict__ Out) {
  __shared__ __align__(16) half_t pl[16 * 32];   // P tile: 16 rows x 32 keys

  const int lane = threadIdx.x & 31;
  const int ln = lane & 15;
  const int hi = lane >> 4;
  const int qb = blockIdx.x;
  const int h  = blockIdx.y;
  const int b  = blockIdx.z;

  const size_t bh = (size_t)b * NHEADS + h;
  const half_t* Qb = Qh + bh * SEQ * HD;
  const half_t* Kb = Kh + bh * SEQ * HD;
  const half_t* Vb = Vt + bh * HD * SEQ;

  F16x16 qa[4];
  {
    const half_t* qrow = Qb + (size_t)(qb * 16 + ln) * HD;
#pragma unroll
    for (int f = 0; f < 4; ++f) {
      qa[f].u[0] = *(const uint4*)(qrow + f * 32 + hi * 8);
      qa[f].u[1] = *(const uint4*)(qrow + f * 32 + 16 + hi * 8);
    }
  }

  F32x8 o[8];
#pragma unroll
  for (int dt = 0; dt < 8; ++dt) o[dt] = F32x8{};
  float mrow[8], lrow[8];
#pragma unroll
  for (int v = 0; v < 8; ++v) { mrow[v] = NEGBIG; lrow[v] = 0.0f; }

  const float scale = 0.08838834764831845f;   // 1/sqrt(128)
  const int kmax = qb * 16 + 16;

  for (int kp = 0; kp < kmax; kp += 32) {
    const bool has2 = (kp + 16) < kmax;       // uniform across the wave

    F32x8 s0 = {}, s1 = {};
#pragma unroll
    for (int f = 0; f < 4; ++f) {
      F16x16 kb;
      const half_t* kr = Kb + (size_t)(kp + ln) * HD + f * 32 + hi * 8;
      kb.u[0] = *(const uint4*)(kr);
      kb.u[1] = *(const uint4*)(kr + 16);
      s0.v = __builtin_amdgcn_wmma_f32_16x16x32_f16(
          false, qa[f].v, false, kb.v, (short)0, s0.v, false, false);
    }
    if (has2) {
#pragma unroll
      for (int f = 0; f < 4; ++f) {
        F16x16 kb;
        const half_t* kr = Kb + (size_t)(kp + 16 + ln) * HD + f * 32 + hi * 8;
        kb.u[0] = *(const uint4*)(kr);
        kb.u[1] = *(const uint4*)(kr + 16);
        s1.v = __builtin_amdgcn_wmma_f32_16x16x32_f16(
            false, qa[f].v, false, kb.v, (short)0, s1.v, false, false);
      }
    }

    float p0[8], p1[8], fac[8];
#pragma unroll
    for (int v = 0; v < 8; ++v) {
      const int row = qb * 16 + v + 8 * hi;
      float x0 = s0.f[v] * scale + ((kp + ln > row) ? MASKNEG : 0.0f);
      float x1 = has2 ? (s1.f[v] * scale + ((kp + 16 + ln > row) ? MASKNEG : 0.0f))
                      : NEGBIG;
      const float rmx  = row_max16(fmaxf(x0, x1));
      const float mnew = fmaxf(mrow[v], rmx);
      fac[v] = __expf(mrow[v] - mnew);
      p0[v]  = __expf(x0 - mnew);
      p1[v]  = has2 ? __expf(x1 - mnew) : 0.0f;
      lrow[v] = lrow[v] * fac[v] + row_sum16(p0[v] + p1[v]);
      mrow[v] = mnew;
    }
#pragma unroll
    for (int dt = 0; dt < 8; ++dt)
#pragma unroll
      for (int v = 0; v < 8; ++v) o[dt].f[v] *= fac[v];

#pragma unroll
    for (int v = 0; v < 8; ++v) {
      const int r = v + 8 * hi;
      pl[r * 32 + ln]      = (half_t)p0[v];
      pl[r * 32 + 16 + ln] = (half_t)p1[v];
    }
    __syncthreads();
    F16x16 pa;
    pa.u[0] = *(const uint4*)(&pl[ln * 32 + hi * 8]);
    pa.u[1] = *(const uint4*)(&pl[ln * 32 + 16 + hi * 8]);
    __syncthreads();

#pragma unroll
    for (int dt = 0; dt < 8; ++dt) {
      F16x16 vb;
      const half_t* vr = Vb + (size_t)(dt * 16 + ln) * SEQ + kp + hi * 8;
      vb.u[0] = *(const uint4*)(vr);
      vb.u[1] = *(const uint4*)(vr + 16);
      o[dt].v = __builtin_amdgcn_wmma_f32_16x16x32_f16(
          false, pa.v, false, vb.v, (short)0, o[dt].v, false, false);
    }
  }

#pragma unroll
  for (int v = 0; v < 8; ++v) {
    const float rl = 1.0f / lrow[v];
    const int s = qb * 16 + v + 8 * hi;
    const size_t base = ((size_t)b * SEQ + s) * DIMX + h * HD;
#pragma unroll
    for (int dt = 0; dt < 8; ++dt)
      Out[base + dt * 16 + ln] = (half_t)(o[dt].f[v] * rl);
  }
}

// ---------------------------------------------------------------------------
// Host-side orchestration (all launches on `stream`; scratch from d_ws only).
// ---------------------------------------------------------------------------
extern "C" void kernel_launch(void* const* d_in, const int* in_sizes, int n_in,
                              void* d_out, int out_size, void* d_ws, size_t ws_size,
                              hipStream_t stream) {
  (void)in_sizes; (void)n_in; (void)out_size; (void)ws_size;

  const float* x    = (const float*)d_in[0];
  const float* wq   = (const float*)d_in[1];
  const float* wk   = (const float*)d_in[2];
  const float* wv   = (const float*)d_in[3];
  const float* wo   = (const float*)d_in[4];
  const float* fcos = (const float*)d_in[5];
  const float* fsin = (const float*)d_in[6];
  // d_in[7] (mask) unused: causal mask applied analytically in attn_fwd.

  char* ws = (char*)d_ws;
  size_t off = 0;
  auto halloc = [&](size_t nelem) -> half_t* {
    half_t* p = (half_t*)(ws + off);
    off += nelem * sizeof(half_t);
    return p;
  };
  half_t* Xh  = halloc((size_t)NTOK * DIMX);
  half_t* Wqh = halloc((size_t)DIMX * DIMX);
  half_t* Wkh = halloc((size_t)DIMX * DIMX);
  half_t* Wvh = halloc((size_t)DIMX * DIMX);
  half_t* Woh = halloc((size_t)DIMX * DIMX);
  half_t* Qt  = halloc((size_t)NTOK * DIMX);   // Q = X Wq^T (pre-RoPE)
  half_t* Kt  = halloc((size_t)NTOK * DIMX);
  half_t* Vt0 = halloc((size_t)NTOK * DIMX);
  half_t* Qp  = halloc((size_t)NTOK * DIMX);   // (B,H,S,HD) post-RoPE
  half_t* Kp  = halloc((size_t)NTOK * DIMX);
  half_t* Vp  = halloc((size_t)NTOK * DIMX);   // (B,H,HD,S)
  half_t* Ao  = halloc((size_t)NTOK * DIMX);   // attention output (NTOK,DIM)

  const dim3 eblk(256);
  cvt_f32_f16<<<2048, eblk, 0, stream>>>(x,  Xh,  NTOK * DIMX);
  cvt_f32_f16<<<2048, eblk, 0, stream>>>(wq, Wqh, DIMX * DIMX);
  cvt_f32_f16<<<2048, eblk, 0, stream>>>(wk, Wkh, DIMX * DIMX);
  cvt_f32_f16<<<2048, eblk, 0, stream>>>(wv, Wvh, DIMX * DIMX);
  cvt_f32_f16<<<2048, eblk, 0, stream>>>(wo, Woh, DIMX * DIMX);

  const dim3 gblk(256);
  const dim3 ggrid(DIMX / NBLK, NTOK / MBLK);   // (32, 32)
  gemm_tdm_nt<<<ggrid, gblk, 0, stream>>>(Xh, Wqh, nullptr, Qt,  NTOK, DIMX, DIMX);
  gemm_tdm_nt<<<ggrid, gblk, 0, stream>>>(Xh, Wkh, nullptr, Kt,  NTOK, DIMX, DIMX);
  gemm_tdm_nt<<<ggrid, gblk, 0, stream>>>(Xh, Wvh, nullptr, Vt0, NTOK, DIMX, DIMX);

  const int nrope = NTOK * DIMX / 2;
  rope_pack<<<(nrope + 255) / 256, eblk, 0, stream>>>(Qt, fcos, fsin, Qp);
  rope_pack<<<(nrope + 255) / 256, eblk, 0, stream>>>(Kt, fcos, fsin, Kp);
  const int nv = NTOK * DIMX;
  v_pack<<<(nv + 255) / 256, eblk, 0, stream>>>(Vt0, Vp);

  const dim3 agrid(SEQ / 16, NHEADS, BSZ);
  attn_fwd<<<agrid, dim3(32), 0, stream>>>(Qp, Kp, Vp, Ao);

  gemm_tdm_nt<<<ggrid, gblk, 0, stream>>>(Ao, Woh, (float*)d_out, nullptr,
                                          NTOK, DIMX, DIMX);
}
